// GCNEdgeClassifier_49795850830535
// MI455X (gfx1250) — matrix operand order repaired and checked by reading, not compiled, not run on hardware
//
#include <hip/hip_runtime.h>
#include <hip/hip_bf16.h>

// ---------------------------------------------------------------------------
// GCN edge classifier for MI455X (gfx1250), wave32 + WMMA f16 path.
//   1) deg/norm via f32 atomics
//   2) h = emb[x] -> f16
//   3) 3x GCN layer: hW = h@W (WMMA), scatter msg -> atomicAdd agg[dst], relu
//   4) edge MLP: ef=[h[src]|h[dst]] staged via GLOBAL_LOAD_ASYNC_TO_LDS_B128,
//      hid = relu(ef@Wm1+b) via WMMA, out = hid@Wm2+b
// Weights are pre-packed into WMMA B-fragment order so every B fragment is a
// single contiguous 32B load per lane (2x global_load_b128).
// ---------------------------------------------------------------------------

typedef __attribute__((ext_vector_type(16))) _Float16 v16h;
typedef __attribute__((ext_vector_type(8)))  _Float16 v8h;
typedef __attribute__((ext_vector_type(8)))  float    v8f;

constexpr int N_ = 100000;   // nodes   (divisible by 16)
constexpr int E_ = 1600000;  // edges   (divisible by 16)
constexpr int D_ = 128;

// ---- WMMA fragment loaders (layouts per CDNA5 ISA §7.12.2) ------

// A fragment, 16x32 f16, row = lane%16.
// element e<8  -> K = kbase + half8 + e        (contiguous 16B run)
// element e>=8 -> K = kbase + 16 + half8 + e-8 (contiguous 16B run)
__device__ __forceinline__ v16h load_a16(const _Float16* rowp, int kbase, int half8) {
  v8h lo = *(const v8h*)(rowp + kbase + half8);
  v8h hi = *(const v8h*)(rowp + kbase + 16 + half8);
  return __builtin_shufflevector(lo, hi, 0, 1, 2, 3, 4, 5, 6, 7,
                                         8, 9, 10, 11, 12, 13, 14, 15);
}

// B fragment from pre-packed weights: one contiguous 32B run per lane.
__device__ __forceinline__ v16h load_b16_packed(const _Float16* packed,
                                                int coltile, int kc, int kcCount,
                                                int lane) {
  const _Float16* p = packed + (size_t)(((coltile * kcCount + kc) * 32) + lane) * 16;
  v8h lo = *(const v8h*)(p);
  v8h hi = *(const v8h*)(p + 8);
  return __builtin_shufflevector(lo, hi, 0, 1, 2, 3, 4, 5, 6, 7,
                                         8, 9, 10, 11, 12, 13, 14, 15);
}

// ---- setup kernels ---------------------------------------------------------

__global__ void k_deg_init(float* deg) {
  int i = blockIdx.x * blockDim.x + threadIdx.x;
  if (i < N_) deg[i] = 1.0f;                       // self-loop
}

__global__ void k_deg_accum(const int* __restrict__ ei, float* deg) {
  int e = blockIdx.x * blockDim.x + threadIdx.x;   // E_ exact
  __hip_atomic_fetch_add(&deg[ei[E_ + e]], 1.0f,
                         __ATOMIC_RELAXED, __HIP_MEMORY_SCOPE_AGENT);
}

__global__ void k_deg_fin(float* deg) {
  int i = blockIdx.x * blockDim.x + threadIdx.x;
  if (i < N_) deg[i] = rsqrtf(deg[i]);             // deg >= 1 always
}

__global__ void k_embed(const int* __restrict__ x, const float* __restrict__ emb,
                        _Float16* __restrict__ h16) {
  int i = blockIdx.x * blockDim.x + threadIdx.x;   // N_*D_ exact
  int node = i >> 7, d = i & 127;
  h16[i] = (_Float16)emb[x[node] * D_ + d];
}

// Convert f32 weight (K x 128, row-major) into f16 WMMA B-fragment order:
//   dst[((coltile*kcCount + kc)*32 + lane)*16 + e] =
//       W[(kc*32 + (lane>=16?8:0) + (e<8?e:e+8))*128 + coltile*16 + (lane&15)]
__global__ void k_pack_b(const float* __restrict__ W, _Float16* __restrict__ dst,
                         int K) {
  const int i = blockIdx.x * blockDim.x + threadIdx.x;  // K*128 exact
  const int kcCount = K >> 5;
  const int e       = i & 15;
  const int lane    = (i >> 4) & 31;
  const int rem     = i >> 9;                 // coltile*kcCount + kc
  const int kc      = rem % kcCount;
  const int coltile = rem / kcCount;
  const int k   = kc * 32 + ((lane >= 16) ? 8 : 0) + ((e < 8) ? e : e + 8);
  const int col = coltile * 16 + (lane & 15);
  dst[i] = (_Float16)W[k * D_ + col];
}

__global__ void k_init_agg(float* __restrict__ agg, const float* __restrict__ b) {
  int i = blockIdx.x * blockDim.x + threadIdx.x;   // N_*D_ exact
  agg[i] = b[i & 127];
}

__global__ void k_relu16(const float* __restrict__ agg, _Float16* __restrict__ h16) {
  int i = blockIdx.x * blockDim.x + threadIdx.x;   // N_*D_ exact
  float v = agg[i];
  h16[i] = (_Float16)(v > 0.0f ? v : 0.0f);
}

// ---- node-feature GEMM: hW = h @ W (N x 128 @ 128 x 128), f16 in/out -------
__global__ void k_node_gemm(const _Float16* __restrict__ h16,
                            const _Float16* __restrict__ wpk,   // packed B
                            _Float16* __restrict__ hw16) {
  const int tid  = threadIdx.x;
  const int lane = tid & 31;
  const int wave = tid >> 5;
  const int rowtile = blockIdx.x * 16;
  const int col     = wave * 16 + (lane & 15);
  const int half8   = (lane >= 16) ? 8 : 0;
  const _Float16* arow = h16 + (size_t)(rowtile + (lane & 15)) * D_;

  v8f c = {};
#pragma unroll
  for (int kc = 0; kc < 4; ++kc) {                 // K = 128 in 4 chunks of 32
    v16h a = load_a16(arow, kc * 32, half8);
    v16h b = load_b16_packed(wpk, wave, kc, 4, lane);
    c = __builtin_amdgcn_wmma_f32_16x16x32_f16(false, a, false, b,
                                               (short)0, c, false, false);
  }
  const int rowoff = (lane >= 16) ? 8 : 0;
#pragma unroll
  for (int r = 0; r < 8; ++r)
    hw16[(size_t)(rowtile + r + rowoff) * D_ + col] = (_Float16)c[r];
}

// ---- message scatter: agg[dst] += norm * hW[src], one wave per edge --------
__global__ void k_scatter(const int* __restrict__ ei, const float* __restrict__ dis,
                          const _Float16* __restrict__ hw16, float* __restrict__ agg) {
  const int gw   = blockIdx.x * 8 + (threadIdx.x >> 5);  // edge id
  const int lane = threadIdx.x & 31;
  int s, t;
  if (gw < E_) { s = ei[gw]; t = ei[E_ + gw]; }
  else         { s = t = gw - E_; }                      // self loop
  const float nrm = dis[s] * dis[t];
  const _Float16* hp = hw16 + (size_t)s * D_ + lane * 4;
  float* ap = agg + (size_t)t * D_ + lane * 4;
#pragma unroll
  for (int i = 0; i < 4; ++i)
    __hip_atomic_fetch_add(ap + i, nrm * (float)hp[i],
                           __ATOMIC_RELAXED, __HIP_MEMORY_SCOPE_AGENT);
}

// ---- edge MLP: out = relu([h_src|h_dst] @ Wm1 + bm1) @ Wm2 + bm2 -----------
__global__ void k_edge_mlp(const _Float16* __restrict__ h16,
                           const int* __restrict__ ei,
                           const _Float16* __restrict__ Wm1pk,  // packed 256x128
                           const float* __restrict__ bm1,       // 128
                           const float* __restrict__ Wm2,       // 128 x 3
                           const float* __restrict__ bm2,       // 3
                           float* __restrict__ out) {           // E x 3
  __shared__ __align__(16) _Float16 ef[16][256];
  __shared__ float hid[16][128];

  const int tid   = threadIdx.x;
  const int ebase = blockIdx.x * 16;

  // Stage ef tile with async global->LDS DMA: thread -> (row, 16-half chunk),
  // 2 x 16B per thread. GV mode: each lane copies 16B from its global address to its LDS address.
  {
    const int row   = tid >> 4;          // 0..15
    const int chunk = tid & 15;          // 0..15
    const int e     = ebase + row;
    const int node  = (chunk < 8) ? ei[e] : ei[E_ + e];
    const int c0    = (chunk & 7) * 16;
    const int dst0  = ((chunk < 8) ? 0 : 128) + c0;
    // Low 32 bits of a generic LDS pointer == LDS byte offset (flat aperture
    // mapping uses addr[31:0] as LDS_ADDR).
    unsigned ldsoff = (unsigned)(unsigned long long)(void*)&ef[row][dst0];
    unsigned long long ga = (unsigned long long)(h16 + (size_t)node * D_ + c0);
    asm volatile("global_load_async_to_lds_b128 %0, %1, off"
                 :: "v"(ldsoff), "v"(ga) : "memory");
    asm volatile("global_load_async_to_lds_b128 %0, %1, off offset:16"
                 :: "v"(ldsoff), "v"(ga) : "memory");
  }
  asm volatile("s_wait_asynccnt 0x0" ::: "memory");
  __syncthreads();

  const int lane   = tid & 31;
  const int wave   = tid >> 5;
  const int col    = wave * 16 + (lane & 15);
  const int half8  = (lane >= 16) ? 8 : 0;
  const int rowoff = (lane >= 16) ? 8 : 0;

  v8f c;
  const float binit = bm1[col];
#pragma unroll
  for (int r = 0; r < 8; ++r) c[r] = binit;

  const _Float16* arow = &ef[lane & 15][0];
#pragma unroll
  for (int kc = 0; kc < 8; ++kc) {                 // K = 256 in 8 chunks of 32
    if (kc < 7)
      __builtin_prefetch(Wm1pk + (size_t)((wave * 8 + kc + 1) * 32 + lane) * 16, 0, 1);
    v16h a = load_a16(arow, kc * 32, half8);
    v16h b = load_b16_packed(Wm1pk, wave, kc, 8, lane);
    c = __builtin_amdgcn_wmma_f32_16x16x32_f16(false, a, false, b,
                                               (short)0, c, false, false);
  }

  // ReLU + park hid tile in LDS.
#pragma unroll
  for (int r = 0; r < 8; ++r) {
    float v = c[r];
    hid[r + rowoff][col] = v > 0.0f ? v : 0.0f;
  }
  __syncthreads();

  // 16 edges x 3 outputs = 48 dot products of length 128.
  if (tid < 48) {
    const int row = tid / 3, j = tid % 3;
    float acc = bm2[j];
#pragma unroll 4
    for (int k = 0; k < 128; ++k) acc += hid[row][k] * Wm2[k * 3 + j];
    out[(size_t)(ebase + row) * 3 + j] = acc;
  }
}

// ---------------------------------------------------------------------------

extern "C" void kernel_launch(void* const* d_in, const int* in_sizes, int n_in,
                              void* d_out, int out_size, void* d_ws, size_t ws_size,
                              hipStream_t stream) {
  const int*   x   = (const int*)d_in[0];
  const int*   ei  = (const int*)d_in[1];
  const float* emb = (const float*)d_in[2];
  const float* W0  = (const float*)d_in[3];
  const float* b0  = (const float*)d_in[4];
  const float* W1  = (const float*)d_in[5];
  const float* b1  = (const float*)d_in[6];
  const float* W2  = (const float*)d_in[7];
  const float* b2  = (const float*)d_in[8];
  const float* Wm1 = (const float*)d_in[9];
  const float* bm1 = (const float*)d_in[10];
  const float* Wm2 = (const float*)d_in[11];
  const float* bm2 = (const float*)d_in[12];
  float* out = (float*)d_out;

  // Workspace carve-up (~103 MB): dis | h16 | hw16 | agg | wpk
  char*  ws  = (char*)d_ws;
  size_t off = 0;
  auto carve = [&](size_t bytes) -> void* {
    void* p = ws + off;
    off += (bytes + 255) & ~(size_t)255;
    return p;
  };
  float*    dis  = (float*)   carve((size_t)N_ * sizeof(float));
  _Float16* h16  = (_Float16*)carve((size_t)N_ * D_ * sizeof(_Float16));
  _Float16* hw16 = (_Float16*)carve((size_t)N_ * D_ * sizeof(_Float16));
  float*    agg  = (float*)   carve((size_t)N_ * D_ * sizeof(float));
  _Float16* wpk  = (_Float16*)carve((size_t)5 * D_ * D_ * sizeof(_Float16));
  (void)ws_size; (void)in_sizes; (void)n_in; (void)out_size;

  const int ND_blocks = (N_ * D_) / 256;           // 50000

  k_deg_init <<<(N_ + 255) / 256, 256, 0, stream>>>(dis);
  k_deg_accum<<<E_ / 256,        256, 0, stream>>>(ei, dis);
  k_deg_fin  <<<(N_ + 255) / 256, 256, 0, stream>>>(dis);
  k_embed    <<<ND_blocks,        256, 0, stream>>>(x, emb, h16);

  // Pack weights into WMMA B-fragment order (fused f32->f16 convert).
  k_pack_b<<<(D_ * D_) / 256,     256, 0, stream>>>(W0,  wpk,                       D_);
  k_pack_b<<<(D_ * D_) / 256,     256, 0, stream>>>(W1,  wpk + (size_t)D_ * D_,     D_);
  k_pack_b<<<(D_ * D_) / 256,     256, 0, stream>>>(W2,  wpk + (size_t)2 * D_ * D_, D_);
  k_pack_b<<<(2 * D_ * D_) / 256, 256, 0, stream>>>(Wm1, wpk + (size_t)3 * D_ * D_, 2 * D_);

  const float* biases[3] = {b0, b1, b2};
  for (int l = 0; l < 3; ++l) {
    k_init_agg <<<ND_blocks,     256, 0, stream>>>(agg, biases[l]);
    k_node_gemm<<<N_ / 16,       256, 0, stream>>>(h16, wpk + (size_t)l * D_ * D_, hw16);
    k_scatter  <<<(E_ + N_) / 8, 256, 0, stream>>>(ei, dis, hw16, agg);
    k_relu16   <<<ND_blocks,     256, 0, stream>>>(agg, h16);
  }

  k_edge_mlp<<<E_ / 16, 256, 0, stream>>>(h16, ei, wpk + (size_t)3 * D_ * D_,
                                          bm1, Wm2, bm2, out);
}
